// NN_24472723653267
// MI455X (gfx1250) — matrix operand order repaired and checked by reading, not compiled, not run on hardware
//
#include <hip/hip_runtime.h>
#include <hip/hip_bf16.h>

typedef __attribute__((ext_vector_type(16))) _Float16     v16h;
typedef __attribute__((ext_vector_type(8)))  float        v8f;
typedef __attribute__((ext_vector_type(8)))  unsigned int v8u;

namespace {

constexpr int kB     = 16384;
constexpr int kL     = 50;
constexpr int kD     = 20;
constexpr int kCH    = 16;
constexpr int kH     = 6;
constexpr int kWaves = 2;            // waves per block -> 64 threads, ~59.8 KB LDS

// ---- per-wave LDS arena: every gather target is zero-padded so fragment
// ---- builds need no bounds checks; reshape scatters run as rolled passes.
struct alignas(16) WaveScratch {
  union {                     // 6144 B, lifetimes disjoint:
    _Float16 xpad[64 * 32];   //   phase 1 in : sample x, padded [64][32]
    _Float16 attn1P[48 * 64]; //   phase 6 in : attn1[p][l] image, stride 64
  } u;
  _Float16 xrP[32 * 70];      // xr image [32][70]; xr[c][l] at c*70+2+l (2-halo)
  _Float16 xeP[64 * 32];      // xe row-major padded [64][32]; data at [l][0..15]
  _Float16 attnPh[64 * 16];   // tanh attention map (f16), rows>=50 garbage unread
  float gi[904];              // GRU gates [50][18] + dump slot @900
  float x4[300];              // GRU outputs [L][6] (flat == torch x4_flat)
  float y[300];
  float xcnn[48];
  float hbuf[8];
  float gh[24];
  float xgru[8];
};
// ---- block-shared weight images, transposed [N][Kpad] so K-pairs are contiguous
struct alignas(16) BlockShared {
  _Float16 WE[16 * 32];     // embedding  B(k,n)=we_w[n][k], k<20
  _Float16 WA[16 * 32];     // attn       B(k,n)=attn_w[n][k], k<16
  _Float16 WG[32 * 32];     // gru_wih    valid n<18,k<16
  _Float16 WC[48 * 96];     // combined conv, K=cin*5+tap, valid k<80
  float    biasC[48];
  WaveScratch wv[kWaves];
};

// ISA 7.12.2 16-bit A 16x32 element K index; kb = 8 for lanes 16-31
__device__ constexpr int a_kidx_c(int kb, int v, int h) {
  return (v < 4) ? (kb + 2 * v + h) : (16 + kb + 2 * (v - 4) + h);
}
// xrP byte-element offset for conv im2col K index (folds at compile time)
__device__ constexpr int conv_off(int k) { return (k / 5) * 70 + (k % 5); }

// Row-contiguous fragment loads (K-pair == one dword in the padded images)
__device__ __forceinline__ v16h frag_a32(const _Float16* row, int lane) {
  const unsigned int* p = (const unsigned int*)row;   // 16 dwords = 32 halves
  const int kb = (lane & 16) ? 4 : 0;
  v8u u;
  #pragma unroll
  for (int v = 0; v < 4; ++v) u[v] = p[kb + v];         // K 0-7 / 8-15
  #pragma unroll
  for (int v = 0; v < 4; ++v) u[4 + v] = p[8 + kb + v]; // K 16-23 / 24-31
  return __builtin_bit_cast(v16h, u);
}
__device__ __forceinline__ v16h frag_b32(const _Float16* col, int lane) {
  const unsigned int* p = (const unsigned int*)col;
  const int kb = (lane & 16) ? 4 : 0;                   // lanes16-31: K8-15,24-31
  v8u u;
  #pragma unroll
  for (int v = 0; v < 8; ++v) u[v] = p[kb + v + ((v >= 4) ? 4 : 0)];
  return __builtin_bit_cast(v16h, u);
}
__device__ __forceinline__ v8f wmma_f16(v16h a, v16h b, v8f c) {
  return __builtin_amdgcn_wmma_f32_16x16x32_f16(false, a, false, b, (short)0, c,
                                                false, false);
}
__device__ __forceinline__ float sigmoidf_(float v) { return 1.0f / (1.0f + __expf(-v)); }

} // namespace

__global__ __launch_bounds__(32 * kWaves, 1)
void fused_model_kernel(const float* __restrict__ x,
                        const float* __restrict__ we_w,  const float* __restrict__ we_b,
                        const float* __restrict__ attn_w, const float* __restrict__ attn_b,
                        const float* __restrict__ conv1_w, const float* __restrict__ conv1_b,
                        const float* __restrict__ conv2_w, const float* __restrict__ conv2_b,
                        const float* __restrict__ conv3_w, const float* __restrict__ conv3_b,
                        const float* __restrict__ gru_wih, const float* __restrict__ gru_whh,
                        const float* __restrict__ gru_bih, const float* __restrict__ gru_bhh,
                        const float* __restrict__ fc3_w,  const float* __restrict__ fc3_b,
                        float* __restrict__ out)
{
  __shared__ BlockShared sh;
  const int tid  = threadIdx.x;
  const int lane = tid & 31;
  const int wv   = tid >> 5;
  const int bidx = blockIdx.x * kWaves + wv;
  WaveScratch& s = sh.wv[wv];
  const int  n    = lane & 15;
  const int  mhi  = (lane & 16) ? 8 : 0;
  const bool hi16 = (lane & 16) != 0;
  const float* xb = x + (size_t)bidx * (kL * kD);
  __builtin_prefetch(xb, 0, 3);          // global_prefetch_b8

  // ===== one-time: build padded transposed weight images (block cooperative) =====
  for (int i = tid; i < 16 * 32; i += 64) {
    int nn = i >> 5, k = i & 31;
    sh.WE[i] = (_Float16)((k < kD) ? we_w[nn * kD + k] : 0.0f);
    sh.WA[i] = (_Float16)((k < kCH) ? attn_w[nn * kCH + k] : 0.0f);
  }
  for (int i = tid; i < 32 * 32; i += 64) {
    int nn = i >> 5, k = i & 31;
    sh.WG[i] = (_Float16)((nn < 18 && k < kCH) ? gru_wih[nn * kCH + k] : 0.0f);
  }
  for (int i = tid; i < 48 * 96; i += 64) {
    int oc = i / 96, k = i - oc * 96;
    float val = 0.0f;
    if (k < 80) {
      int cin = k / 5, tap = k - cin * 5;
      if (oc < 16)      { if (tap == 2)             val = conv1_w[oc * kCH + cin]; }
      else if (oc < 32) { if (tap >= 1 && tap <= 3) val = conv2_w[((oc - 16) * kCH + cin) * 3 + tap - 1]; }
      else              {                           val = conv3_w[((oc - 32) * kCH + cin) * 5 + tap]; }
    }
    sh.WC[i] = (_Float16)val;
  }
  if (tid < 48) sh.biasC[tid] = (tid < 16) ? conv1_b[tid]
                              : (tid < 32) ? conv2_b[tid - 16] : conv3_b[tid - 32];
  __syncthreads();

  // ===== per-wave: zero the padded f16 images (contiguous region), stage x =====
  {
    unsigned int* zp = (unsigned int*)s.u.xpad;
    constexpr int kZeroDwords = (sizeof(s.u) + sizeof(s.xrP) + sizeof(s.xeP)) / 4; // 3680
    for (int i = lane; i < kZeroDwords; i += 32) zp[i] = 0u;
    for (int e = lane; e < kL * kD; e += 32) {
      float f = xb[e];
      int l = e / kD, k = e - l * kD;
      s.u.xpad[l * 32 + k] = (_Float16)f;
    }
  }
  __builtin_amdgcn_wave_barrier();

  // ===== Phase 1: embedding  xe = X @ We^T + b  (4 WMMA) =====
  {
    v16h bE = frag_b32(&sh.WE[n * 32], lane);
    float bias = we_b[n];
    const int wbase = mhi * 32 + n;            // affine writeback, no div
    #pragma unroll
    for (int mt = 0; mt < 4; ++mt) {
      v16h a = frag_a32(&s.u.xpad[(mt * 16 + (lane & 15)) * 32], lane);
      v8f c = {};
      c = wmma_f16(a, bE, c);
      #pragma unroll
      for (int r = 0; r < 8; ++r)
        s.xeP[(mt * 16 + r) * 32 + wbase] = (_Float16)(c[r] + bias);
    }
  }
  __builtin_amdgcn_wave_barrier();

  // ---- rolled re-layout: xe (row-major) -> xr image [CH][L] with 2-halo ----
  for (int e = lane; e < kL * kCH; e += 32) {
    _Float16 hv = s.xeP[(e >> 4) * 32 + (e & 15)];
    unsigned cin = (unsigned)e / 50u, pos = (unsigned)e - cin * 50u;
    s.xrP[cin * 70 + 2 + pos] = hv;
  }
  __builtin_amdgcn_wave_barrier();

  // ===== Phase 2: attn = tanh(xt @ Wa^T + b), xt[l][c]=xr[c][l]  (4 WMMA) =====
  {
    v16h bA = frag_b32(&sh.WA[n * 32], lane);
    float bias = attn_b[n];
    #pragma unroll
    for (int mt = 0; mt < 4; ++mt) {
      int l0 = mt * 16 + (lane & 15);
      v16h a;
      #pragma unroll
      for (int v = 0; v < 8; ++v)
        #pragma unroll
        for (int h = 0; h < 2; ++h) {
          int off = hi16 ? (a_kidx_c(8, v, h) * 70 + 2)    // constexpr-folded
                         : (a_kidx_c(0, v, h) * 70 + 2);
          a[2 * v + h] = s.xrP[off + l0];
        }
      v8f c = {};
      c = wmma_f16(a, bA, c);
      #pragma unroll
      for (int r = 0; r < 8; ++r)
        s.attnPh[(mt * 16 + r + mhi) * kCH + n] = (_Float16)tanhf(c[r] + bias);
    }
  }
  __builtin_amdgcn_wave_barrier();

  // ---- rolled re-layout: attn map -> attn1[p][l] image (stride 64) ----
  for (int e = lane; e < 48 * kL; e += 32) {      // e == flat == p*50+l
    unsigned row = (unsigned)e / 48u;
    unsigned col = ((unsigned)e - row * 48u) & 15u;
    unsigned p = (unsigned)e / 50u, l = (unsigned)e - p * 50u;
    s.u.attn1P[p * 64 + l] = s.attnPh[row * kCH + col];
  }
  __builtin_amdgcn_wave_barrier();

  // ===== Phase 3: GRU input gates  gi = xe @ Wih^T + bih  (8 WMMA) =====
  #pragma unroll
  for (int nt = 0; nt < 2; ++nt) {
    int g = nt * 16 + n;
    v16h bG = frag_b32(&sh.WG[g * 32], lane);
    float bias = gru_bih[(g < 18) ? g : 0];
    #pragma unroll
    for (int mt = 0; mt < 4; ++mt) {
      v16h a = frag_a32(&s.xeP[(mt * 16 + (lane & 15)) * 32], lane);
      v8f c = {};
      c = wmma_f16(a, bG, c);
      #pragma unroll
      for (int r = 0; r < 8; ++r) {
        int l = mt * 16 + r + mhi;
        int off = (l < kL && g < 18) ? (l * 18 + g) : 900;   // address-select
        s.gi[off] = c[r] + bias;
      }
    }
  }
  __builtin_amdgcn_wave_barrier();

  // ===== Phase 4: GRU sequential scan (50 steps, VALU) =====
  {
    int gg = (lane < 18) ? lane : 0;
    float wh[kH], bh = gru_bhh[gg];
    #pragma unroll
    for (int k = 0; k < kH; ++k) wh[k] = gru_whh[gg * kH + k];
    if (lane < kH) s.hbuf[lane] = 0.0f;
    __builtin_amdgcn_wave_barrier();
    for (int t = 0; t < kL; ++t) {
      float g = bh;
      #pragma unroll
      for (int k = 0; k < kH; ++k) g += s.hbuf[k] * wh[k];
      if (lane < 18) s.gh[lane] = g;
      __builtin_amdgcn_wave_barrier();
      if (lane < kH) {
        float r = sigmoidf_(s.gi[t * 18 + lane] + s.gh[lane]);
        float z = sigmoidf_(s.gi[t * 18 + 6 + lane] + s.gh[6 + lane]);
        float nn = tanhf(s.gi[t * 18 + 12 + lane] + r * s.gh[12 + lane]);
        float hn = (1.0f - z) * nn + z * s.hbuf[lane];
        s.hbuf[lane] = hn;
        s.x4[t * kH + lane] = hn;
      }
      __builtin_amdgcn_wave_barrier();
    }
  }

  // ===== Phase 5: self-attention over x4, softmax over dim=1 (shfl reductions) =====
  for (int idx = lane; idx < kL * kH; idx += 32) s.y[idx] = 0.0f;
  __builtin_amdgcn_wave_barrier();
  {
    const int  i1   = lane + 32;
    const bool has1 = (i1 < kL);
    const int  i1c  = has1 ? i1 : 0;
    for (int j = 0; j < kL; ++j) {
      float sc0 = 0.0f, sc1 = 0.0f;
      #pragma unroll
      for (int h = 0; h < kH; ++h) {
        float rv = s.x4[h * kL + j];            // R[h][j] = x4_flat[h*50+j]
        sc0 += s.x4[lane * kH + h] * rv;
        sc1 += s.x4[i1c * kH + h] * rv;
      }
      sc0 *= (1.0f / kH);
      sc1 *= (1.0f / kH);
      float m = fmaxf(sc0, has1 ? sc1 : -3.0e38f);
      #pragma unroll
      for (int d = 16; d; d >>= 1) m = fmaxf(m, __shfl_xor(m, d, 32));
      float e0 = __expf(sc0 - m);
      float e1 = has1 ? __expf(sc1 - m) : 0.0f;
      float se = e0 + e1;
      #pragma unroll
      for (int d = 16; d; d >>= 1) se += __shfl_xor(se, d, 32);
      float inv = 1.0f / se;
      #pragma unroll
      for (int h = 0; h < kH; ++h) {
        float xv = s.x4[j * kH + h];
        s.y[lane * kH + h] += e0 * inv * xv;
        if (has1) s.y[i1 * kH + h] += e1 * inv * xv;
      }
      __builtin_amdgcn_wave_barrier();
    }
  }
  if (lane < kH) {                               // x_gru[k] = sum y_flat[50k..50k+49]
    float sg = 0.0f;
    for (int t = 0; t < kL; ++t) sg += s.y[lane * kL + t];
    s.xgru[lane] = sg;
  }
  __builtin_amdgcn_wave_barrier();

  // ===== Phase 6: fused conv GEMM [50x80]@[80x48] + lrelu + attn1-weighted sum (36 WMMA) =====
  if (lane < 32) s.xcnn[lane] = 0.0f;
  if (lane < 16) s.xcnn[32 + lane] = 0.0f;
  __builtin_amdgcn_wave_barrier();
  {
    v16h bC[3][3];
    #pragma unroll
    for (int nt = 0; nt < 3; ++nt)
      #pragma unroll
      for (int kt = 0; kt < 3; ++kt)
        bC[nt][kt] = frag_b32(&sh.WC[(nt * 16 + n) * 96 + kt * 32], lane);
    float partial[3] = {0.0f, 0.0f, 0.0f};
    #pragma unroll
    for (int mt = 0; mt < 4; ++mt) {
      int l0 = mt * 16 + (lane & 15);
      v16h a[3];                                  // reused across the 3 N-tiles
      #pragma unroll
      for (int kt = 0; kt < 3; ++kt)
        #pragma unroll
        for (int v = 0; v < 8; ++v)
          #pragma unroll
          for (int h = 0; h < 2; ++h) {
            // constexpr-folded offsets; 2-way select on lane-half, no division
            int off = hi16 ? conv_off(kt * 32 + a_kidx_c(8, v, h))
                           : conv_off(kt * 32 + a_kidx_c(0, v, h));
            a[kt][2 * v + h] = s.xrP[off + l0];   // halo handles tap-2
          }
      #pragma unroll
      for (int nt = 0; nt < 3; ++nt) {
        v8f acc = {};
        #pragma unroll
        for (int kt = 0; kt < 3; ++kt) acc = wmma_f16(a[kt], bC[nt][kt], acc);
        float bias = sh.biasC[nt * 16 + n];
        const int abase = (nt * 16 + n) * 64 + mhi + mt * 16;  // attn1P row base
        #pragma unroll
        for (int r = 0; r < 8; ++r) {
          int l = mt * 16 + r + mhi;
          float av = (float)s.u.attn1P[abase + r];  // in-range; stale if l>=50
          float vv = acc[r] + bias;
          vv = (vv > 0.0f) ? vv : 0.1f * vv;        // leaky relu
          if (l < kL) partial[nt] += vv * av;       // pure-VALU select
        }
      }
    }
    #pragma unroll
    for (int nt = 0; nt < 3; ++nt) atomicAdd(&s.xcnn[nt * 16 + n], partial[nt]);
  }
  __builtin_amdgcn_wave_barrier();

  // ===== Phase 7: head =====
  if (lane < 2) {
    float o = fc3_b[lane];
    for (int q = 0; q < 48; ++q) o += s.xcnn[q] * fc3_w[lane * 54 + q];
    #pragma unroll
    for (int q = 0; q < kH; ++q) o += s.xgru[q] * fc3_w[lane * 54 + 48 + q];
    out[(size_t)bidx * 2 + lane] = o;
  }
}

extern "C" void kernel_launch(void* const* d_in, const int* in_sizes, int n_in,
                              void* d_out, int out_size, void* d_ws, size_t ws_size,
                              hipStream_t stream) {
  (void)in_sizes; (void)n_in; (void)out_size; (void)d_ws; (void)ws_size;
  const float* x       = (const float*)d_in[0];
  const float* we_w    = (const float*)d_in[1];
  const float* we_b    = (const float*)d_in[2];
  const float* attn_w  = (const float*)d_in[3];
  const float* attn_b  = (const float*)d_in[4];
  const float* conv1_w = (const float*)d_in[5];
  const float* conv1_b = (const float*)d_in[6];
  const float* conv2_w = (const float*)d_in[7];
  const float* conv2_b = (const float*)d_in[8];
  const float* conv3_w = (const float*)d_in[9];
  const float* conv3_b = (const float*)d_in[10];
  const float* gru_wih = (const float*)d_in[11];
  const float* gru_whh = (const float*)d_in[12];
  const float* gru_bih = (const float*)d_in[13];
  const float* gru_bhh = (const float*)d_in[14];
  const float* fc3_w   = (const float*)d_in[15];
  const float* fc3_b   = (const float*)d_in[16];
  float* out = (float*)d_out;

  dim3 grid(kB / kWaves);          // 8192 blocks, one sample per wave
  dim3 block(32 * kWaves);         // 64 threads = 2 wave32
  fused_model_kernel<<<grid, block, 0, stream>>>(
      x, we_w, we_b, attn_w, attn_b, conv1_w, conv1_b, conv2_w, conv2_b,
      conv3_w, conv3_b, gru_wih, gru_whh, gru_bih, gru_bhh, fc3_w, fc3_b, out);
}
